// SelfAttnV2E_43336220017260
// MI455X (gfx1250) — matrix operand order repaired and checked by reading, not compiled, not run on hardware
//
#include <hip/hip_runtime.h>
#include <hip/hip_bf16.h>

// ---------------------------------------------------------------------------
// MI455X (gfx1250): bf16 WMMA (16x16x32) for all dense GEMMs, f32 VALU for
// LN / softmax / gathers. wave32, 256-thread blocks (8 waves).
// Weights are pre-swizzled into WMMA B-fragment order; the per-K-step weight
// slab is double-buffered in LDS and staged with GLOBAL_LOAD_ASYNC_TO_LDS_B128
// (ASYNCcnt), so the copy overlaps the WMMAs of the current step with no VGPR
// round trip. s_wait_asynccnt 0 + workgroup barrier publish each slab.
// ---------------------------------------------------------------------------

typedef __attribute__((ext_vector_type(16))) __bf16         v16bf;
typedef __attribute__((ext_vector_type(8)))  float          v8f;
typedef __attribute__((ext_vector_type(8)))  unsigned short ushort8;

union ABfrag  { ushort8 h[2]; v16bf v; };

__device__ __forceinline__ unsigned short f2bf(float f) {
  union { float f; unsigned int u; } a; a.f = f;
  unsigned int u = a.u;
  unsigned int r = u + 0x7FFFu + ((u >> 16) & 1u);   // round-to-nearest-even
  return (unsigned short)(r >> 16);
}

__device__ __forceinline__ float gelu_f(float x) {
  return 0.5f * x * (1.0f + erff(x * 0.70710678118654752f));
}

// Async global->LDS 16-byte copy (per lane). dst must be an LDS pointer:
// generic LDS pointers carry the LDS byte offset in their low 32 bits.
__device__ __forceinline__ void async_copy_b128(void* lds_dst, const void* gsrc) {
  const unsigned lds_off = (unsigned)(unsigned long long)lds_dst;
  asm volatile("global_load_async_to_lds_b128 %0, %1, off"
               :: "v"(lds_off), "v"(gsrc)
               : "memory");
}
__device__ __forceinline__ void wait_async0() {
  asm volatile("s_wait_asynccnt 0x0" ::: "memory");
}

// ---------------------------------------------------------------------------
// WMMA GEMM inner loop: 16 rows x (NT*128) cols, K = Krows (multiple of 32).
// 8 waves; wave w owns col tiles [w*NT, (w+1)*NT).
// Asm: LDS A tile, row-major [16][astride] bf16 (astride multiple of 8).
// Wswz: global weights pre-swizzled per (k-step, col-tile) B-fragment tile:
//   element ((ks*ntc + nt)*32 + lane)*16 + e  ==  W[ks*32 + (lane>>4)*16 + e]
//                                                 [nt*16 + (lane&15)]
// sB: LDS double buffer [2][32*CT] halfs.
// ---------------------------------------------------------------------------
template <int NT>
__device__ __forceinline__ void wmma_gemmT(const unsigned short* __restrict__ Asm,
                                           int astride,
                                           const unsigned short* __restrict__ Wswz,
                                           int Krows,
                                           unsigned short* __restrict__ sB,
                                           int tid, v8f* acc) {
  constexpr int CT     = NT * 128;   // total output columns
  constexpr int SLAB   = 32 * CT;    // halfs per k-slab
  constexpr int CHUNK8 = SLAB / 8;   // 16B chunks per slab
  const int wave = tid >> 5;
  const int lane = tid & 31;
  const int row  = lane & 15;        // A-frag row (M)
  const int kb   = (lane >> 4) * 16; // A-frag K sub-offset
  const int nsteps = Krows >> 5;

  auto stage = [&](int ks, int bufi) {
    const unsigned short* src = Wswz + (size_t)ks * SLAB;
    unsigned short* dst = sB + bufi * SLAB;
#pragma unroll
    for (int i = 0; i < CHUNK8 / 256; ++i) {
      const int c = (i * 256 + tid) * 8;
      async_copy_b128(dst + c, src + c);           // 16B/lane, coalesced
    }
  };

  stage(0, 0);
  wait_async0();
  __syncthreads();
  for (int ks = 0; ks < nsteps; ++ks) {
    if (ks + 1 < nsteps) stage(ks + 1, (ks + 1) & 1);   // overlaps WMMAs below
    const unsigned short* buf = sB + (ks & 1) * SLAB;
    ABfrag af;                                      // 2 x ds_load_b128
    const ushort8* ap = (const ushort8*)(Asm + row * astride + ks * 32 + kb);
    af.h[0] = ap[0];
    af.h[1] = ap[1];
#pragma unroll
    for (int t = 0; t < NT; ++t) {
      ABfrag bf;                                    // 2 x ds_load_b128
      const ushort8* bp =
          (const ushort8*)(buf + ((wave * NT + t) * 32 + lane) * 16);
      bf.h[0] = bp[0];
      bf.h[1] = bp[1];
      acc[t] = __builtin_amdgcn_wmma_f32_16x16x32_bf16(
          false, af.v, false, bf.v, (short)0, acc[t], false, false);
    }
    wait_async0();                                  // next slab landed in LDS
    __syncthreads();
  }
}

// ---------------------------------------------------------------------------
// Fused residual MLP block:
//   xin = a[row] (+ pre_add)                      (residual kept in f32)
//   z   = [ LN(xin)*g+b  ||  pe_row ]             (pe only when K1 == 512)
//   h   = gelu(z @ W1 + b1)                       (WMMA bf16)
//   y   = xin + h @ W2 + b2 (+ bias_row)          (WMMA bf16)
// One block = 16 rows; 256 threads.
// ---------------------------------------------------------------------------
__global__ __launch_bounds__(256)
void mlp_block_kernel(const float* __restrict__ a,
                      const float* __restrict__ pre_add,
                      const float* __restrict__ ln_g, const float* __restrict__ ln_b,
                      const unsigned short* __restrict__ W1, const float* __restrict__ b1,
                      int K1,
                      const unsigned short* __restrict__ W2, const float* __restrict__ b2,
                      const float* __restrict__ pe_table, const int* __restrict__ pe_idx,
                      int pe_fixed,
                      const float* __restrict__ bias_table, const int* __restrict__ bias_idx,
                      int bias_fixed,
                      float* __restrict__ out, int M) {
  __shared__ float          sX[16][256];
  __shared__ unsigned short sA[16][512];
  __shared__ unsigned short sH[16][256];
  __shared__ unsigned short sB[2 * 32 * 256];
  __shared__ float red1[16][16];
  __shared__ float red2[16][16];
  __shared__ float rmean[16], rrstd[16];

  const int tid   = threadIdx.x;
  const int wave  = tid >> 5;
  const int lane  = tid & 31;
  const int r0    = blockIdx.x * 16;
  const int row16 = tid >> 4;         // 0..15 (LN row)
  const int seg16 = (tid & 15) * 16;  // 16 cols per thread

  // ---- load input tile (+ optional broadcast add), LN partial sums ----
  {
    const int grow = r0 + row16;
    float s1 = 0.f, s2 = 0.f;
    for (int i = 0; i < 16; ++i) {
      const int c = seg16 + i;
      float v = (grow < M) ? a[(size_t)grow * 256 + c] : 0.f;
      if (pre_add) v += pre_add[c];
      sX[row16][c] = v;
      s1 += v; s2 += v * v;
    }
    red1[row16][tid & 15] = s1;
    red2[row16][tid & 15] = s2;
  }
  __syncthreads();
  if ((tid & 15) == 0) {
    float s1 = 0.f, s2 = 0.f;
    for (int i = 0; i < 16; ++i) { s1 += red1[row16][i]; s2 += red2[row16][i]; }
    const float m   = s1 * (1.0f / 256.0f);
    const float var = s2 * (1.0f / 256.0f) - m * m;
    rmean[row16] = m;
    rrstd[row16] = rsqrtf(var + 1e-5f);
  }
  __syncthreads();
  // ---- normalized A tile (bf16), optional PE concat ----
  {
    const int grow = r0 + row16;
    const float m  = rmean[row16];
    const float rs = rrstd[row16];
    const float* perow = nullptr;
    if (K1 == 512) {
      const int pidx = pe_idx ? ((grow < M) ? pe_idx[grow] : 0) : pe_fixed;
      perow = pe_table + (size_t)pidx * 256;
    }
    for (int i = 0; i < 16; ++i) {
      const int c = seg16 + i;
      const float v = (sX[row16][c] - m) * rs * ln_g[c] + ln_b[c];
      sA[row16][c] = f2bf(v);
      if (K1 == 512) sA[row16][256 + c] = f2bf(perow[c]);
    }
  }
  __syncthreads();

  // ---- GEMM1: [16,K1] x [K1,256] -> gelu -> sH ----
  {
    v8f acc[2] = {};
    wmma_gemmT<2>(&sA[0][0], 512, W1, K1, sB, tid, acc);
    const int coll  = lane & 15;
    const int mhalf = (lane >> 4) ? 8 : 0;
#pragma unroll
    for (int t = 0; t < 2; ++t) {
      const int c = wave * 32 + t * 16 + coll;
#pragma unroll
      for (int g = 0; g < 8; ++g) {
        const int mrow = g + mhalf;
        sH[mrow][c] = f2bf(gelu_f(acc[t][g] + b1[c]));
      }
    }
  }
  __syncthreads();

  // ---- GEMM2: [16,256] x [256,256] + residual (+ bias row) ----
  {
    v8f acc[2] = {};
    wmma_gemmT<2>(&sH[0][0], 256, W2, 256, sB, tid, acc);
    const int coll  = lane & 15;
    const int mhalf = (lane >> 4) ? 8 : 0;
#pragma unroll
    for (int t = 0; t < 2; ++t) {
      const int c = wave * 32 + t * 16 + coll;
#pragma unroll
      for (int g = 0; g < 8; ++g) {
        const int mrow = g + mhalf;
        const int grow = r0 + mrow;
        if (grow < M) {
          float o = acc[t][g] + b2[c] + sX[mrow][c];
          if (bias_table) {
            const int bidx = bias_idx ? bias_idx[grow] : bias_fixed;
            o += bias_table[(size_t)bidx * 256 + c];
          }
          out[(size_t)grow * 256 + c] = o;
        }
      }
    }
  }
}

// ---------------------------------------------------------------------------
// k/v projections + attention logits, fused per 16-row tile.
//   k = x1 @ kW + kb   [.,512]  (kept in LDS for logit dot products)
//   v = x1 @ vW + vb   [.,256]  (written to global)
//   l{0,1}[n,h] = (1/sqrt(32)) * dot(q{0,1}[h], k{0,1}[n,h])
// ---------------------------------------------------------------------------
__global__ __launch_bounds__(256)
void qkv_kernel(const float* __restrict__ x1,
                const unsigned short* __restrict__ kW, const float* __restrict__ kb,
                const unsigned short* __restrict__ vW, const float* __restrict__ vb,
                const float* __restrict__ qbuf,
                float* __restrict__ vout, float* __restrict__ l0, float* __restrict__ l1,
                int M) {
  __shared__ unsigned short sA[16 * 256];
  __shared__ unsigned short sB[2 * 32 * 512];
  __shared__ float          sK[16 * 512];

  const int tid  = threadIdx.x;
  const int wave = tid >> 5;
  const int lane = tid & 31;
  const int r0   = blockIdx.x * 16;

  {
    const int row = tid >> 4, cseg = (tid & 15) * 16, grow = r0 + row;
    for (int i = 0; i < 16; ++i) {
      const float v = (grow < M) ? x1[(size_t)grow * 256 + cseg + i] : 0.f;
      sA[row * 256 + cseg + i] = f2bf(v);
    }
  }
  __syncthreads();

  // k projection: 512 output cols (4 tiles per wave)
  {
    v8f acc[4] = {};
    wmma_gemmT<4>(sA, 256, kW, 256, sB, tid, acc);
    const int coll  = lane & 15;
    const int mhalf = (lane >> 4) ? 8 : 0;
#pragma unroll
    for (int t = 0; t < 4; ++t) {
      const int c = wave * 64 + t * 16 + coll;
#pragma unroll
      for (int g = 0; g < 8; ++g)
        sK[(g + mhalf) * 512 + c] = acc[t][g] + kb[c];
    }
  }
  __syncthreads();

  // v projection: 256 output cols -> global
  {
    v8f acc[2] = {};
    wmma_gemmT<2>(sA, 256, vW, 256, sB, tid, acc);
    const int coll  = lane & 15;
    const int mhalf = (lane >> 4) ? 8 : 0;
#pragma unroll
    for (int t = 0; t < 2; ++t) {
      const int c = wave * 32 + t * 16 + coll;
#pragma unroll
      for (int g = 0; g < 8; ++g) {
        const int grow = r0 + g + mhalf;
        if (grow < M) vout[(size_t)grow * 256 + c] = acc[t][g] + vb[c];
      }
    }
  }
  __syncthreads();

  // logits: 16 rows x 8 heads x 2 orders = 256 dot products of length 32
  {
    const int row = tid >> 4;
    const int rem = tid & 15;
    const int ord = rem >> 3;
    const int h   = rem & 7;
    const int grow = r0 + row;
    float s = 0.f;
    for (int d = 0; d < 32; ++d)
      s += qbuf[ord * 256 + h * 32 + d] * sK[row * 512 + ord * 256 + h * 32 + d];
    s *= 0.17677669529663687f;  // 1/sqrt(32)
    if (grow < M) (ord ? l1 : l0)[(size_t)grow * 8 + h] = s;
  }
}

// ---------------------------------------------------------------------------
// order-0 softmax stats over all N rows (max + sum-exp per head), and zero
// the att0 accumulator.
// ---------------------------------------------------------------------------
__global__ __launch_bounds__(256)
void softmax0_stats_kernel(const float* __restrict__ l0, float* __restrict__ stats,
                           float* __restrict__ att0raw, int M) {
  __shared__ float red[256];
  const int tid = threadIdx.x;
  const int h   = tid & 7;
  float pm = -3.4e38f;
  for (int n = tid >> 3; n < M; n += 32) pm = fmaxf(pm, l0[(size_t)n * 8 + h]);
  red[tid] = pm;
  __syncthreads();
  for (int s = 128; s >= 8; s >>= 1) {
    if (tid < s) red[tid] = fmaxf(red[tid], red[tid + s]);
    __syncthreads();
  }
  const float mx = red[h];
  __syncthreads();
  float ps = 0.f;
  for (int n = tid >> 3; n < M; n += 32) ps += expf(l0[(size_t)n * 8 + h] - mx);
  red[tid] = ps;
  __syncthreads();
  for (int s = 128; s >= 8; s >>= 1) {
    if (tid < s) red[tid] += red[tid + s];
    __syncthreads();
  }
  if (tid < 8) { stats[tid] = mx; stats[8 + tid] = red[tid]; }
  att0raw[tid] = 0.f;
}

// att0[h*32+d] = sum_n softmax0(n,h) * v[n, h*32+d]   (grid-stride + atomics)
__global__ __launch_bounds__(256)
void att0_accum_kernel(const float* __restrict__ l0, const float* __restrict__ v,
                       const float* __restrict__ stats, float* __restrict__ att0raw,
                       int M) {
  const int c = threadIdx.x;
  const int h = c >> 5;
  const float mx  = stats[h];
  const float inv = 1.0f / stats[8 + h];
  float acc = 0.f;
  for (int n = blockIdx.x; n < M; n += gridDim.x)
    acc += expf(l0[(size_t)n * 8 + h] - mx) * inv * v[(size_t)n * 256 + c];
  atomicAdd(&att0raw[c], acc);
}

// ---------------------------------------------------------------------------
// order-1 edge attention: one block per edge; its 8 incident entries are
// contiguous (edge_idx = repeat(arange(E), 8)). Segment softmax over 8,
// weighted gather-sum of v rows (the dominant HBM gather, ~820 MB).
// ---------------------------------------------------------------------------
__global__ __launch_bounds__(256)
void edge_att_kernel(const float* __restrict__ l1, const float* __restrict__ v,
                     const int* __restrict__ node_idx, float* __restrict__ e1, int E) {
  __shared__ int   nid[8];
  __shared__ float lg[8][8];
  __shared__ float alpha[8][8];
  __shared__ float mh[8], dh[8];
  const int e   = blockIdx.x;
  const int tid = threadIdx.x;
  if (tid < 8) nid[tid] = node_idx[(size_t)e * 8 + tid];
  __syncthreads();
  if (tid < 64) { const int j = tid >> 3, h = tid & 7; lg[j][h] = l1[(size_t)nid[j] * 8 + h]; }
  __syncthreads();
  if (tid < 8) {
    const int h = tid;
    float m = -3.4e38f;
    for (int j = 0; j < 8; ++j) m = fmaxf(m, lg[j][h]);
    float d = 0.f;
    for (int j = 0; j < 8; ++j) d += expf(lg[j][h] - m);
    mh[h] = m; dh[h] = d;
  }
  __syncthreads();
  if (tid < 64) { const int j = tid >> 3, h = tid & 7; alpha[j][h] = expf(lg[j][h] - mh[h]) / dh[h]; }
  __syncthreads();
  const int h = tid >> 5;
  float acc = 0.f;
  for (int j = 0; j < 8; ++j) acc += alpha[j][h] * v[(size_t)nid[j] * 256 + tid];
  e1[(size_t)e * 256 + tid] = acc;
}

// ---------------------------------------------------------------------------
// Tiny precompute: sinusoidal PE tables, PositionalMLP queries, bias rows.
// ---------------------------------------------------------------------------
__global__ __launch_bounds__(256)
void prep_small_kernel(const float* __restrict__ qW1, const float* __restrict__ qb1,
                       const float* __restrict__ qW2, const float* __restrict__ qb2,
                       const float* __restrict__ bW1, const float* __restrict__ bb1,
                       const float* __restrict__ bW2, const float* __restrict__ bb2,
                       float* __restrict__ qbuf, float* __restrict__ ball,
                       float* __restrict__ pe2t, float* __restrict__ pe3t) {
  __shared__ float pe64[9 * 64];
  __shared__ float hid[9 * 256];
  const int tid = threadIdx.x;
  const float nl = -logf(10000.0f);
  for (int idx = tid; idx < 9 * 64; idx += 256) {
    const int p = idx >> 6, j = idx & 63;
    const float div = expf(nl * (float)(j & ~1) / 64.0f);
    const float ang = (float)p * div;
    pe64[idx] = (j & 1) ? cosf(ang) : sinf(ang);
  }
  for (int idx = tid; idx < 9 * 256; idx += 256) {
    const int p = idx >> 8, j = idx & 255;
    const float div = expf(nl * (float)(j & ~1) / 256.0f);
    const float ang = (float)p * div;
    const float val = (j & 1) ? cosf(ang) : sinf(ang);
    pe3t[idx] = val;
    if (p < 2) pe2t[idx] = val;
  }
  __syncthreads();
  for (int idx = tid; idx < 2 * 256; idx += 256) {      // q hidden
    const int r = idx >> 8, c = idx & 255;
    float s = qb1[c];
    for (int i = 0; i < 64; ++i) s += pe64[r * 64 + i] * qW1[i * 256 + c];
    hid[idx] = gelu_f(s);
  }
  __syncthreads();
  for (int idx = tid; idx < 2 * 256; idx += 256) {      // q out [2,256]
    const int r = idx >> 8, c = idx & 255;
    float s = qb2[c];
    for (int i = 0; i < 256; ++i) s += hid[r * 256 + i] * qW2[i * 256 + c];
    qbuf[idx] = s;
  }
  __syncthreads();
  for (int idx = tid; idx < 9 * 256; idx += 256) {      // bias hidden
    const int r = idx >> 8, c = idx & 255;
    float s = bb1[c];
    for (int i = 0; i < 64; ++i) s += pe64[r * 64 + i] * bW1[i * 256 + c];
    hid[idx] = gelu_f(s);
  }
  __syncthreads();
  for (int idx = tid; idx < 9 * 256; idx += 256) {      // b_all [9,256]
    const int r = idx >> 8, c = idx & 255;
    float s = bb2[c];
    for (int i = 0; i < 256; ++i) s += hid[r * 256 + i] * bW2[i * 256 + c];
    ball[idx] = s;
  }
}

// ---------------------------------------------------------------------------
// Convert f32 weight [K][C] into bf16 WMMA B-fragment order:
//   out[((ks*(C/16) + nt)*32 + l)*16 + e] = W[ks*32 + (l>>4)*16 + e][nt*16 + (l&15)]
// ---------------------------------------------------------------------------
__global__ __launch_bounds__(256)
void swizzle_w_kernel(const float* __restrict__ W, unsigned short* __restrict__ out,
                      int K, int C) {
  const int idx = blockIdx.x * 256 + threadIdx.x;
  if (idx >= K * C) return;
  const int e    = idx & 15;
  const int l    = (idx >> 4) & 31;
  const int tile = idx >> 9;         // ks * (C/16) + nt
  const int ntc  = C >> 4;
  const int ks   = tile / ntc;
  const int nt   = tile - ks * ntc;
  const int col  = nt * 16 + (l & 15);
  const int kk   = ks * 32 + ((l >> 4) * 16) + e;
  out[idx] = f2bf(W[(size_t)kk * C + col]);
}

// ---------------------------------------------------------------------------
extern "C" void kernel_launch(void* const* d_in, const int* in_sizes, int n_in,
                              void* d_out, int out_size, void* d_ws, size_t ws_size,
                              hipStream_t stream) {
  (void)n_in; (void)out_size; (void)ws_size;
  const float* x           = (const float*)d_in[0];
  const int*   node_idx    = (const int*)d_in[1];
  /* d_in[2] edge_idx: structurally repeat(arange(E),8) -> contiguous segments */
  const int*   edge_orders = (const int*)d_in[3];
  const float* qW1 = (const float*)d_in[4];
  const float* qb1 = (const float*)d_in[5];
  const float* qW2 = (const float*)d_in[6];
  const float* qb2 = (const float*)d_in[7];
  const float* kW  = (const float*)d_in[8];
  const float* kb  = (const float*)d_in[9];
  const float* vW  = (const float*)d_in[10];
  const float* vb  = (const float*)d_in[11];
  const float* m1W1 = (const float*)d_in[12];
  const float* m1b1 = (const float*)d_in[13];
  const float* m1W2 = (const float*)d_in[14];
  const float* m1b2 = (const float*)d_in[15];
  const float* m2W1 = (const float*)d_in[16];
  const float* m2b1 = (const float*)d_in[17];
  const float* m2W2 = (const float*)d_in[18];
  const float* m2b2 = (const float*)d_in[19];
  const float* m3W1 = (const float*)d_in[20];
  const float* m3b1 = (const float*)d_in[21];
  const float* m3W2 = (const float*)d_in[22];
  const float* m3b2 = (const float*)d_in[23];
  const float* n1g = (const float*)d_in[24];
  const float* n1b = (const float*)d_in[25];
  const float* n2g = (const float*)d_in[26];
  const float* n2b = (const float*)d_in[27];
  const float* n3g = (const float*)d_in[28];
  const float* n3b = (const float*)d_in[29];
  const float* bW1 = (const float*)d_in[30];
  const float* bb1 = (const float*)d_in[31];
  const float* bW2 = (const float*)d_in[32];
  const float* bb2 = (const float*)d_in[33];

  const int N = in_sizes[0] / 256;
  const int E = in_sizes[3];

  char* ws = (char*)d_ws;
  size_t off = 0;
  auto alloc = [&](size_t bytes) {
    void* p = ws + off;
    off = (off + bytes + 255) & ~(size_t)255;
    return p;
  };
  unsigned short* kWb   = (unsigned short*)alloc((size_t)256 * 512 * 2);
  unsigned short* vWb   = (unsigned short*)alloc((size_t)256 * 256 * 2);
  unsigned short* m1W1b = (unsigned short*)alloc((size_t)256 * 256 * 2);
  unsigned short* m1W2b = (unsigned short*)alloc((size_t)256 * 256 * 2);
  unsigned short* m2W1b = (unsigned short*)alloc((size_t)512 * 256 * 2);
  unsigned short* m2W2b = (unsigned short*)alloc((size_t)256 * 256 * 2);
  unsigned short* m3W1b = (unsigned short*)alloc((size_t)512 * 256 * 2);
  unsigned short* m3W2b = (unsigned short*)alloc((size_t)256 * 256 * 2);
  float* x1      = (float*)alloc((size_t)N * 256 * 4);
  float* vbuf    = (float*)alloc((size_t)N * 256 * 4);
  float* l0      = (float*)alloc((size_t)N * 8 * 4);
  float* l1      = (float*)alloc((size_t)N * 8 * 4);
  float* e1      = (float*)alloc((size_t)E * 256 * 4);
  float* qbuf    = (float*)alloc(512 * 4);
  float* ball    = (float*)alloc(9 * 256 * 4);
  float* pe2t    = (float*)alloc(2 * 256 * 4);
  float* pe3t    = (float*)alloc(9 * 256 * 4);
  float* att0raw = (float*)alloc(256 * 4);
  float* att0f   = (float*)alloc(256 * 4);
  float* stats   = (float*)alloc(16 * 4);

  auto swz = [&](const float* s, unsigned short* d, int K, int C) {
    swizzle_w_kernel<<<(K * C + 255) / 256, 256, 0, stream>>>(s, d, K, C);
  };
  swz(kW, kWb, 256, 512);
  swz(vW, vWb, 256, 256);
  swz(m1W1, m1W1b, 256, 256);
  swz(m1W2, m1W2b, 256, 256);
  swz(m2W1, m2W1b, 512, 256);
  swz(m2W2, m2W2b, 256, 256);
  swz(m3W1, m3W1b, 512, 256);
  swz(m3W2, m3W2b, 256, 256);

  prep_small_kernel<<<1, 256, 0, stream>>>(qW1, qb1, qW2, qb2, bW1, bb1, bW2, bb2,
                                           qbuf, ball, pe2t, pe3t);

  const int gN = (N + 15) / 16, gE = (E + 15) / 16;

  // block 1: x1 = x + mlp(ln(x))
  mlp_block_kernel<<<gN, 256, 0, stream>>>(x, nullptr, n1g, n1b,
                                           m1W1b, m1b1, 256, m1W2b, m1b2,
                                           nullptr, nullptr, 0, nullptr, nullptr, 0,
                                           x1, N);
  // k/v projections + logits
  qkv_kernel<<<gN, 256, 0, stream>>>(x1, kWb, kb, vWb, vb, qbuf, vbuf, l0, l1, N);
  // order-0 attention
  softmax0_stats_kernel<<<1, 256, 0, stream>>>(l0, stats, att0raw, N);
  att0_accum_kernel<<<256, 256, 0, stream>>>(l0, vbuf, stats, att0raw, N);
  // order-1 edge attention
  edge_att_kernel<<<E, 256, 0, stream>>>(l1, vbuf, node_idx, e1, E);
  // blk2 (pe2): att0, att1_v (in-place on vbuf), att1_e (in-place on e1)
  mlp_block_kernel<<<1, 256, 0, stream>>>(att0raw, nullptr, n2g, n2b,
                                          m2W1b, m2b1, 512, m2W2b, m2b2,
                                          pe2t, nullptr, 0, nullptr, nullptr, 0,
                                          att0f, 1);
  mlp_block_kernel<<<gN, 256, 0, stream>>>(vbuf, nullptr, n2g, n2b,
                                           m2W1b, m2b1, 512, m2W2b, m2b2,
                                           pe2t, nullptr, 1, nullptr, nullptr, 0,
                                           vbuf, N);
  mlp_block_kernel<<<gE, 256, 0, stream>>>(e1, nullptr, n2g, n2b,
                                           m2W1b, m2b1, 512, m2W2b, m2b2,
                                           pe2t, nullptr, 1, nullptr, nullptr, 0,
                                           e1, E);
  // blk3 (pe3) with x = att0 + att1_*, plus final BiasE row -> outputs
  float* out_v = (float*)d_out;
  float* out_e = (float*)d_out + (size_t)N * 256;
  mlp_block_kernel<<<gN, 256, 0, stream>>>(vbuf, att0f, n3g, n3b,
                                           m3W1b, m3b1, 512, m3W2b, m3b2,
                                           pe3t, nullptr, 1, ball, nullptr, 1,
                                           out_v, N);
  mlp_block_kernel<<<gE, 256, 0, stream>>>(e1, att0f, n3g, n3b,
                                           m3W1b, m3b1, 512, m3W2b, m3b2,
                                           pe3t, edge_orders, 0, ball, edge_orders, 0,
                                           out_e, E);
}
